// StateLeakyAssocModel_79388175499707
// MI455X (gfx1250) — matrix operand order repaired and checked by reading, not compiled, not run on hardware
//
#include <hip/hip_runtime.h>

// ---------------------------------------------------------------------------
// Model dims (fixed per reference)
// ---------------------------------------------------------------------------
#define T_LEN   2048
#define BATCH   32
#define IN_DIM  64
#define H_DIM   256
#define OUT_DIM 63
#define ROWS    (T_LEN * BATCH)     // 65536 (row = t*B + b, matches x layout)
#define CHANNELS (BATCH * H_DIM)    // 8192  (channel = b*H + h)
#define CHUNK   32
#define NCHUNK  (T_LEN / CHUNK)     // 64
#define BETA    0.9f
#define BETA32  0.034336838202925124f  // 0.9^32

typedef __attribute__((ext_vector_type(16))) __bf16 v16bf;
typedef __attribute__((ext_vector_type(8)))  __bf16 v8bf;
typedef __attribute__((ext_vector_type(4)))  float  v4f;
typedef __attribute__((ext_vector_type(8)))  float  v8f;

union AFrag { v16bf v; v8bf h[2]; };

// ---------------------------------------------------------------------------
// Prep 1: x fp32 -> bf16 (vectorized, 4 elems/thread)
// ---------------------------------------------------------------------------
__global__ __launch_bounds__(256)
void cvt_x_kernel(const float* __restrict__ in, __bf16* __restrict__ out) {
  const size_t i = ((size_t)blockIdx.x * 256 + threadIdx.x) * 4;
  const v4f f = *(const v4f*)(in + i);
  __bf16 b0 = (__bf16)f.x, b1 = (__bf16)f.y, b2 = (__bf16)f.z, b3 = (__bf16)f.w;
  __bf16* o = out + i;
  o[0] = b0; o[1] = b1; o[2] = b2; o[3] = b3;
}

// ---------------------------------------------------------------------------
// Prep 2: pack W (Kdim x Nreal fp32, row-major) into per-lane WMMA B-fragment
// order, bf16, zero-padded. Output index == tid with nesting [nt][ks][lane][e]:
//   lane = hi*16+lm holds N = nt*16+lm, element e holds K = ks*32 + hi*16 + e.
// Each GEMM lane then loads one contiguous 32-byte fragment per K-step.
// ---------------------------------------------------------------------------
__global__ __launch_bounds__(256)
void pack_w_kernel(const float* __restrict__ W, __bf16* __restrict__ out,
                   int Kdim, int Nreal, int NKT) {
  const int tid  = blockIdx.x * 256 + threadIdx.x;
  const int e    = tid & 15;
  const int lane = (tid >> 4) & 31;
  const int ks   = (tid >> 9) % NKT;
  const int nt   = (tid >> 9) / NKT;
  const int hi = lane >> 4, lm = lane & 15;
  const int k = ks * 32 + hi * 16 + e;
  const int n = nt * 16 + lm;
  const float v = (k < Kdim && n < Nreal) ? W[(size_t)k * Nreal + n] : 0.0f;
  out[tid] = (__bf16)v;
}

// ---------------------------------------------------------------------------
// Shared GEMM core: one wave computes a 16x16 tile; block = 8 waves = 64x32.
// A: row-major bf16 (lda = K). Wpk: packed fragments, NKT K-tiles per N-tile.
// ---------------------------------------------------------------------------
template <int K>
static __device__ __forceinline__ v8f wmma_tile(const __bf16* __restrict__ A,
                                                const __bf16* __restrict__ Wpk,
                                                int row, int ntile, int lane) {
  const int hi = lane >> 4;
  constexpr int NKT = K / 32;
  const __bf16* bp = Wpk + ((size_t)ntile * NKT * 32 + lane) * 16;
  v8f acc = {};
#pragma unroll
  for (int ks = 0; ks < NKT; ++ks) {
    AFrag a, b;
    const __bf16* ap = A + (size_t)row * K + ks * 32;
    a.h[0] = *(const v8bf*)(ap + hi * 16);       // K = ks*32 + hi*16 + [0..7]
    a.h[1] = *(const v8bf*)(ap + hi * 16 + 8);   // K = ks*32 + hi*16 + 8 + [0..7]
    b.v = *(const v16bf*)(bp + (size_t)ks * 512);
    acc = __builtin_amdgcn_wmma_f32_16x16x32_bf16(false, a.v, false, b.v,
                                                  (short)0, acc, false, false);
  }
  return acc;
}

// GEMM1: h1 = xb * W_in + b_in (K=64)
__global__ __launch_bounds__(256)
void gemm_in_kernel(const __bf16* __restrict__ xb, const __bf16* __restrict__ Wpk,
                    const float* __restrict__ bias, __bf16* __restrict__ h1) {
  const int lane = threadIdx.x & 31, wave = threadIdx.x >> 5;
  const int hi = lane >> 4, lm = lane & 15;
  const int tileM = blockIdx.x * 64 + (wave & 3) * 16;
  const int ntile = blockIdx.y * 2 + (wave >> 2);
  const int row = tileM + lm, col = ntile * 16 + lm;
  const v8f acc = wmma_tile<IN_DIM>(xb, Wpk, row, ntile, lane);
  const float bn = bias[col];
#pragma unroll
  for (int v = 0; v < 8; ++v)
    h1[(size_t)(tileM + hi * 8 + v) * H_DIM + col] = (__bf16)(acc[v] + bn);
}

// GEMM2: h2 = relu(h1s * W_mid + b_mid) (K=256)
__global__ __launch_bounds__(256)
void gemm_mid_kernel(const __bf16* __restrict__ A, const __bf16* __restrict__ Wpk,
                     const float* __restrict__ bias, __bf16* __restrict__ out) {
  const int lane = threadIdx.x & 31, wave = threadIdx.x >> 5;
  const int hi = lane >> 4, lm = lane & 15;
  const int tileM = blockIdx.x * 64 + (wave & 3) * 16;
  const int ntile = blockIdx.y * 2 + (wave >> 2);
  const int row = tileM + lm, col = ntile * 16 + lm;
  const v8f acc = wmma_tile<H_DIM>(A, Wpk, row, ntile, lane);
  const float bn = bias[col];
#pragma unroll
  for (int v = 0; v < 8; ++v)
    out[(size_t)(tileM + hi * 8 + v) * H_DIM + col] = (__bf16)fmaxf(acc[v] + bn, 0.0f);
}

// GEMM3: y = sigmoid(h2s * W_out + b_out), N padded 63->64 in packed weights
__global__ __launch_bounds__(256)
void gemm_out_kernel(const __bf16* __restrict__ A, const __bf16* __restrict__ Wpk,
                     const float* __restrict__ bias, float* __restrict__ y) {
  const int lane = threadIdx.x & 31, wave = threadIdx.x >> 5;
  const int hi = lane >> 4, lm = lane & 15;
  const int tileM = blockIdx.x * 64 + (wave & 3) * 16;
  const int ntile = blockIdx.y * 2 + (wave >> 2);
  const int row = tileM + lm, col = ntile * 16 + lm;   // 0..63; 63 is pad
  const v8f acc = wmma_tile<H_DIM>(A, Wpk, row, ntile, lane);
  if (col < OUT_DIM) {
    const float bn = bias[col];
#pragma unroll
    for (int v = 0; v < 8; ++v) {
      const float s = acc[v] + bn;
      y[(size_t)(tileM + hi * 8 + v) * OUT_DIM + col] = 1.0f / (1.0f + __expf(-s));
    }
  }
}

// ---------------------------------------------------------------------------
// Chunked leaky scan over T. h stored bf16 as (T*B, H); stride_t = 8192.
// ---------------------------------------------------------------------------
__global__ __launch_bounds__(256)
void scan_local_kernel(__bf16* __restrict__ h, float* __restrict__ carry) {
  const int c = blockIdx.y * blockDim.x + threadIdx.x;   // channel 0..8191
  const int j = blockIdx.x;                              // chunk 0..63
  const size_t base = (size_t)j * CHUNK * CHANNELS + c;
  float s = 0.0f;
#pragma unroll 4
  for (int l = 0; l < CHUNK; ++l) {
    const size_t idx = base + (size_t)l * CHANNELS;
    s = BETA * s + (float)h[idx];
    h[idx] = (__bf16)s;
  }
  carry[(size_t)c * NCHUNK + j] = s;
}

__global__ __launch_bounds__(256)
void scan_carry_kernel(float* __restrict__ carry) {
  const int c = blockIdx.x * blockDim.x + threadIdx.x;
  float* p = carry + (size_t)c * NCHUNK;
  float E = 0.0f;
#pragma unroll 4
  for (int j = 0; j < NCHUNK; ++j) {
    const float cj = p[j];
    p[j] = E;                        // exclusive prefix entering chunk j
    E = BETA32 * E + cj;
  }
}

__global__ __launch_bounds__(256)
void scan_apply_kernel(__bf16* __restrict__ h, const float* __restrict__ prefix) {
  const int c = blockIdx.y * blockDim.x + threadIdx.x;
  const int j = blockIdx.x;
  const float E = prefix[(size_t)c * NCHUNK + j];
  const size_t base = (size_t)j * CHUNK * CHANNELS + c;
  float f = BETA;
#pragma unroll 4
  for (int l = 0; l < CHUNK; ++l) {
    const size_t idx = base + (size_t)l * CHANNELS;
    h[idx] = (__bf16)((float)h[idx] + f * E);
    f *= BETA;
  }
}

// ---------------------------------------------------------------------------
extern "C" void kernel_launch(void* const* d_in, const int* in_sizes, int n_in,
                              void* d_out, int out_size, void* d_ws, size_t ws_size,
                              hipStream_t stream) {
  const float* x     = (const float*)d_in[0];
  const float* W_in  = (const float*)d_in[1];
  const float* b_in  = (const float*)d_in[2];
  const float* W_mid = (const float*)d_in[3];
  const float* b_mid = (const float*)d_in[4];
  const float* W_out = (const float*)d_in[5];
  const float* b_out = (const float*)d_in[6];
  float* y = (float*)d_out;

  char* ws = (char*)d_ws;
  size_t off = 0;
  __bf16* h1    = (__bf16*)(ws + off); off += (size_t)ROWS * H_DIM * 2;   // 32 MiB
  __bf16* h2    = (__bf16*)(ws + off); off += (size_t)ROWS * H_DIM * 2;   // 32 MiB
  float*  carry = (float*)(ws + off);  off += (size_t)CHANNELS * NCHUNK * 4; // 2 MiB
  __bf16* xb    = (__bf16*)(ws + off); off += (size_t)ROWS * IN_DIM * 2;  // 8 MiB
  __bf16* Wp_in  = (__bf16*)(ws + off); off += (size_t)16 * 2 * 512 * 2;  // 32 KiB
  __bf16* Wp_mid = (__bf16*)(ws + off); off += (size_t)16 * 8 * 512 * 2;  // 128 KiB
  __bf16* Wp_out = (__bf16*)(ws + off); off += (size_t)4 * 8 * 512 * 2;   // 32 KiB

  const dim3 blk(256);
  const dim3 gemm_grid(ROWS / 64, H_DIM / 32);     // 1024 x 8
  const dim3 gemm3_grid(ROWS / 64, 64 / 32);       // 1024 x 2
  const dim3 scan_grid(NCHUNK, CHANNELS / 256);    // 64 x 32
  const dim3 carry_grid(CHANNELS / 256);           // 32

  // Prep: convert x, pack weights (tiny, L2-resident)
  cvt_x_kernel<<<dim3((ROWS * IN_DIM) / (256 * 4)), blk, 0, stream>>>(x, xb);
  pack_w_kernel<<<dim3(16 * 2 * 512 / 256), blk, 0, stream>>>(W_in,  Wp_in,  IN_DIM, H_DIM, 2);
  pack_w_kernel<<<dim3(16 * 8 * 512 / 256), blk, 0, stream>>>(W_mid, Wp_mid, H_DIM, H_DIM, 8);
  pack_w_kernel<<<dim3(4 * 8 * 512 / 256),  blk, 0, stream>>>(W_out, Wp_out, H_DIM, OUT_DIM, 8);

  gemm_in_kernel<<<gemm_grid, blk, 0, stream>>>(xb, Wp_in, b_in, h1);
  scan_local_kernel<<<scan_grid, blk, 0, stream>>>(h1, carry);
  scan_carry_kernel<<<carry_grid, blk, 0, stream>>>(carry);
  scan_apply_kernel<<<scan_grid, blk, 0, stream>>>(h1, carry);

  gemm_mid_kernel<<<gemm_grid, blk, 0, stream>>>(h1, Wp_mid, b_mid, h2);
  scan_local_kernel<<<scan_grid, blk, 0, stream>>>(h2, carry);
  scan_carry_kernel<<<carry_grid, blk, 0, stream>>>(carry);
  scan_apply_kernel<<<scan_grid, blk, 0, stream>>>(h2, carry);

  gemm_out_kernel<<<gemm3_grid, blk, 0, stream>>>(h2, Wp_out, b_out, y);
}
